// tf_ContextKnowledgeEncoder_54339926229501
// MI455X (gfx1250) — compile-verified, matrix operand order checked by reading, not run on hardware
//
#include <hip/hip_runtime.h>
#include <hip/hip_bf16.h>

// ---------------------------------------------------------------------------
// ContextKnowledgeEncoder forward for MI455X (gfx1250, wave32, WMMA).
// Shapes: B=16, Ts=256, K=32, Tk=128, D=512, H=8, depth=64, DFF=2048, L=2.
// Token rows: ctx rows [0,4096), knowledge rows [4096, 69632).
// Dense GEMMs: v_wmma_f32_16x16x32_f16, 128x128 block tile, 8 WMMAs per wave
// per K-step, double-buffered LDS in WMMA fragment layout (v16h loads =
// 2x ds_load_b128), register-prefetch of the next tile to overlap HBM/L2
// latency with the matrix pipe.
// ---------------------------------------------------------------------------

typedef __attribute__((ext_vector_type(16))) _Float16 v16h;
typedef __attribute__((ext_vector_type(8)))  float    v8f;

#define NTOK      69632        // 16*256 + 512*128
#define CTX_ROWS  4096
#define DMODEL    512
#define DFF_N     2048
#define BM 128
#define BN 128
#define BK 32

// ---------------- weight conversion f32 -> f16 -----------------------------
__global__ __launch_bounds__(256)
void f32_to_f16_kernel(const float* __restrict__ s, _Float16* __restrict__ d, int n) {
  int i = blockIdx.x * 256 + threadIdx.x;
  if (i < n) d[i] = (_Float16)s[i];
}

// ---------------- embedding + positional + pad mask ------------------------
__global__ __launch_bounds__(256)
void embed_kernel(const int* __restrict__ src, const int* __restrict__ know,
                  const float* __restrict__ emb, const float* __restrict__ pos,
                  float* __restrict__ X, float* __restrict__ maskf) {
  int row = blockIdx.x;
  int tok, t;
  if (row < CTX_ROWS) { t = row & 255; tok = src[row]; }
  else                { int r = row - CTX_ROWS; t = r & 127; tok = know[r]; }
  const float scale = 22.627416997969522f;   // sqrt(512)
  for (int d = threadIdx.x; d < DMODEL; d += 256)
    X[(size_t)row * DMODEL + d] =
        emb[(size_t)tok * DMODEL + d] * scale + pos[(size_t)t * DMODEL + d];
  if (threadIdx.x == 0) maskf[row] = (tok == 0) ? 1.0f : 0.0f;
}

// ---------------- WMMA GEMM: C = relu?(A[f32] x W[f16] + bias) -------------
// A: M x Kdim (f32, row-major), W: Kdim x N (f16, row-major), C: M x N (f32).
// Block = 256 threads = 8 waves, tile 128x128. Wave (s0 = w&3, nh = w>>2)
// owns rows {s0*16.., s0*16+64..} x cols {nh*64..}: 2 A-frags, 4 B-frags,
// 8 WMMAs per K-step. LDS tiles are double buffered in fragment layout.
__global__ __launch_bounds__(256)
void gemm_wmma_f16(const float* __restrict__ A, const _Float16* __restrict__ W,
                   const float* __restrict__ bias, float* __restrict__ C,
                   int M, int N, int Kdim, int relu) {
  __shared__ v16h As2[2][8][32];   // [buf][m-strip][lane]  (16 KB)
  __shared__ v16h Bs2[2][8][32];   // [buf][n-tile][lane]   (16 KB)

  const int tid  = threadIdx.x;
  const int lane = tid & 31;
  const int wave = tid >> 5;          // 0..7
  const int s0   = wave & 3;          // A strips s0 and s0+4
  const int nt0  = (wave >> 2) * 4;   // 4 owned n-tiles
  const int bm = blockIdx.x * BM;
  const int bn = blockIdx.y * BN;

  v8f acc0[4] = {v8f{}, v8f{}, v8f{}, v8f{}};
  v8f acc1[4] = {v8f{}, v8f{}, v8f{}, v8f{}};

  float    aLo[8], aHi[8];   // prefetched A pairs (f32)
  unsigned bPk[8];           // prefetched B pairs (2x f16 packed)

  // Issue global loads for the tile at k-offset kk into registers.
  auto loadRegs = [&](int kk) {
#pragma unroll
    for (int j = 0; j < 8; ++j) {               // A: 128x32 = 2048 pairs
      int i = tid + j * 256;
      int r = i >> 4, c = (i & 15) * 2;
      const float* ap = &A[(size_t)(bm + r) * Kdim + (kk + c)];
      aLo[j] = ap[0];
      aHi[j] = ap[1];
    }
#pragma unroll
    for (int j = 0; j < 8; ++j) {               // B: 32x128 = 2048 k-pairs
      int i = tid + j * 256;
      int r = (i >> 7) * 2, c = i & 127;
      union { _Float16 h[2]; unsigned u; } pk;
      pk.h[0] = W[(size_t)(kk + r)     * N + (bn + c)];
      pk.h[1] = W[(size_t)(kk + r + 1) * N + (bn + c)];
      bPk[j] = pk.u;
    }
  };
  // Convert/store prefetched registers into LDS buffer sb, fragment-swizzled:
  // A (r,c) -> As2[sb][r>>4][(r&15)|(((c>>3)&1)<<4)] slot ((c>>4)<<3)|(c&7)
  // B (k,c) -> Bs2[sb][c>>4][(c&15)|(((k>>3)&1)<<4)] slot ((k>>4)<<3)|(k&7)
  auto storeRegs = [&](int sb) {
#pragma unroll
    for (int j = 0; j < 8; ++j) {
      int i = tid + j * 256;
      int r = i >> 4, c = (i & 15) * 2;
      union { _Float16 h[2]; unsigned u; } pk;
      pk.h[0] = (_Float16)aLo[j];
      pk.h[1] = (_Float16)aHi[j];
      int ln = (r & 15) | (((c >> 3) & 1) << 4);
      int e  = ((c >> 4) << 3) | (c & 7);
      *(unsigned*)((_Float16*)&As2[sb][r >> 4][ln] + e) = pk.u;
    }
#pragma unroll
    for (int j = 0; j < 8; ++j) {
      int i = tid + j * 256;
      int r = (i >> 7) * 2, c = i & 127;
      int ln = (c & 15) | (((r >> 3) & 1) << 4);
      int e  = ((r >> 4) << 3) | (r & 7);
      *(unsigned*)((_Float16*)&Bs2[sb][c >> 4][ln] + e) = bPk[j];
    }
  };

  loadRegs(0);
  storeRegs(0);
  int buf = 0;
  for (int k0 = 0; k0 < Kdim; k0 += BK) {
    __syncthreads();                       // LDS[buf] ready; buf^1 drained
    const bool next = (k0 + BK) < Kdim;
    if (next) loadRegs(k0 + BK);           // global loads in flight over WMMAs

    v16h a0 = As2[buf][s0][lane];          // 2x ds_load_b128 each
    v16h a1 = As2[buf][s0 + 4][lane];
#pragma unroll
    for (int t = 0; t < 4; ++t) {
      v16h b = Bs2[buf][nt0 + t][lane];
      acc0[t] = __builtin_amdgcn_wmma_f32_16x16x32_f16(false, a0, false, b,
                                                       (short)0, acc0[t],
                                                       false, false);
      acc1[t] = __builtin_amdgcn_wmma_f32_16x16x32_f16(false, a1, false, b,
                                                       (short)0, acc1[t],
                                                       false, false);
    }
    if (next) storeRegs(buf ^ 1);
    buf ^= 1;
  }

  // Epilogue: C layout — lanes 0-15: M=r, lanes 16-31: M=8+r; N = lane&15.
  const int mrow0 = bm + s0 * 16 + ((lane >> 4) << 3);
  const int mrow1 = mrow0 + 64;
  const int ncb   = bn + (wave >> 2) * 64 + (lane & 15);
#pragma unroll
  for (int t = 0; t < 4; ++t) {
    const int nc = ncb + t * 16;
    const float bv = bias[nc];
#pragma unroll
    for (int r = 0; r < 8; ++r) {
      float v0 = acc0[t][r] + bv;
      float v1 = acc1[t][r] + bv;
      if (relu) { v0 = fmaxf(v0, 0.0f); v1 = fmaxf(v1, 0.0f); }
      C[(size_t)(mrow0 + r) * N + nc] = v0;
      C[(size_t)(mrow1 + r) * N + nc] = v1;
    }
  }
}

// ---------------- softmax attention: one block per (seq, head, query) ------
__global__ __launch_bounds__(64)
void attn_kernel(const float* __restrict__ Q, const float* __restrict__ Kb,
                 const float* __restrict__ V, const float* __restrict__ maskf,
                 float* __restrict__ O, int T, int rowbase) {
  __shared__ float qv[64];
  __shared__ float sc[256];
  __shared__ float red[64];
  const int tid = threadIdx.x;
  const int q   = blockIdx.x % T;
  const int h   = (blockIdx.x / T) & 7;
  const int s   = blockIdx.x / (T * 8);
  const size_t base = (size_t)rowbase + (size_t)s * T;
  const size_t qrow = base + q;
  const int hb = h * 64;

  qv[tid] = Q[qrow * DMODEL + hb + tid];
  __syncthreads();

  float lmax = -1e30f;
  for (int k = tid; k < T; k += 64) {
    const size_t krow = base + k;
    float dot = 0.0f;
    for (int d = 0; d < 64; ++d) dot += qv[d] * Kb[krow * DMODEL + hb + d];
    float v = dot * 0.125f - 1e9f * maskf[krow];   // /sqrt(64) + pad mask
    sc[k] = v;
    lmax = fmaxf(lmax, v);
  }
  red[tid] = lmax; __syncthreads();
  for (int off = 32; off > 0; off >>= 1) {
    if (tid < off) red[tid] = fmaxf(red[tid], red[tid + off]);
    __syncthreads();
  }
  const float mx = red[0];
  __syncthreads();

  float lsum = 0.0f;
  for (int k = tid; k < T; k += 64) {
    float e = __expf(sc[k] - mx);
    sc[k] = e;
    lsum += e;
  }
  red[tid] = lsum; __syncthreads();
  for (int off = 32; off > 0; off >>= 1) {
    if (tid < off) red[tid] += red[tid + off];
    __syncthreads();
  }
  const float inv = 1.0f / red[0];
  __syncthreads();

  float acc = 0.0f;
  for (int k = 0; k < T; ++k)
    acc += sc[k] * V[(base + k) * DMODEL + hb + tid];
  O[qrow * DMODEL + hb + tid] = acc * inv;
}

// ---------------- residual + LayerNorm (in place over X) -------------------
__global__ __launch_bounds__(64)
void ln_residual_kernel(float* __restrict__ X, const float* __restrict__ F,
                        const float* __restrict__ g, const float* __restrict__ b) {
  __shared__ float rs[64], rq[64];
  const int tid = threadIdx.x;
  const size_t row = blockIdx.x;
  float v[8];
  float s = 0.0f, sq = 0.0f;
#pragma unroll
  for (int i = 0; i < 8; ++i) {
    int c = tid + i * 64;
    float x = X[row * DMODEL + c] + F[row * DMODEL + c];
    v[i] = x; s += x; sq += x * x;
  }
  rs[tid] = s; rq[tid] = sq; __syncthreads();
  for (int off = 32; off > 0; off >>= 1) {
    if (tid < off) { rs[tid] += rs[tid + off]; rq[tid] += rq[tid + off]; }
    __syncthreads();
  }
  const float mean = rs[0] * (1.0f / 512.0f);
  const float var  = rq[0] * (1.0f / 512.0f) - mean * mean;
  const float rstd = rsqrtf(var + 1e-6f);
#pragma unroll
  for (int i = 0; i < 8; ++i) {
    int c = tid + i * 64;
    X[row * DMODEL + c] = g[c] * (v[i] - mean) * rstd + b[c];
  }
}

// ---------------- universal sentence embedding -----------------------------
__global__ __launch_bounds__(128)
void use_emb_kernel(const float* __restrict__ X, const float* __restrict__ maskf,
                    float* __restrict__ U, int T, int rowbase) {
  const int sidx = blockIdx.x;
  const int tid  = threadIdx.x;
  float acc[4] = {0.f, 0.f, 0.f, 0.f};
  float len = 0.0f;
  for (int t = 0; t < T; ++t) {
    const size_t row = (size_t)rowbase + (size_t)sidx * T + t;
    const float m = 1.0f - maskf[row];
    len += m;
#pragma unroll
    for (int i = 0; i < 4; ++i)
      acc[i] += X[row * DMODEL + tid + i * 128] * m;
  }
  const float inv = (1.0f / (sqrtf(len) + 1e-6f)) * 0.044194173824159216f; // /sqrt(512)
#pragma unroll
  for (int i = 0; i < 4; ++i)
    U[(size_t)sidx * DMODEL + tid + i * 128] = acc[i] * inv;
}

// ---------------- ck_attn[b,k] = <know_use[b,k,:], ctx_use[b,:]> -----------
__global__ __launch_bounds__(64)
void ck_attn_kernel(const float* __restrict__ ctxU, const float* __restrict__ knowU,
                    float* __restrict__ ck) {
  __shared__ float red[64];
  const int tid = threadIdx.x;
  const int b = blockIdx.x >> 5;
  float s = 0.0f;
#pragma unroll
  for (int i = 0; i < 8; ++i) {
    int d = tid + i * 64;
    s += knowU[(size_t)blockIdx.x * DMODEL + d] * ctxU[(size_t)b * DMODEL + d];
  }
  red[tid] = s; __syncthreads();
  for (int off = 32; off > 0; off >>= 1) {
    if (tid < off) red[tid] += red[tid + off];
    __syncthreads();
  }
  if (tid == 0) ck[blockIdx.x] = red[0];
}

// ---------------- final gather/concat of full_enc + full_mask --------------
__global__ __launch_bounds__(256)
void assemble_kernel(const float* __restrict__ X, const float* __restrict__ maskf,
                     const int* __restrict__ chosen,
                     float* __restrict__ out_enc, float* __restrict__ out_mask) {
  const int bt = blockIdx.x;     // 0 .. 16*384-1
  const int b = bt / 384;
  const int t = bt % 384;
  size_t srow;
  if (t < 128) {                 // chosen knowledge segment (use_ext == 1.0)
    const int ck = chosen[b];
    srow = (size_t)CTX_ROWS + ((size_t)(b * 32 + ck)) * 128 + t;
  } else {                       // context segment
    srow = (size_t)b * 256 + (t - 128);
  }
  for (int i = threadIdx.x; i < DMODEL; i += 256)
    out_enc[(size_t)bt * DMODEL + i] = X[srow * DMODEL + i];
  if (threadIdx.x == 0) out_mask[bt] = maskf[srow];
}

// ---------------------------------------------------------------------------
extern "C" void kernel_launch(void* const* d_in, const int* in_sizes, int n_in,
                              void* d_out, int out_size, void* d_ws, size_t ws_size,
                              hipStream_t stream) {
  (void)in_sizes; (void)n_in; (void)out_size; (void)ws_size;

  const int*   src    = (const int*)d_in[0];     // [16,256]
  const int*   know   = (const int*)d_in[1];     // [16,32,128]
  const int*   chosen = (const int*)d_in[2];     // [16,1]
  const float* emb    = (const float*)d_in[3];   // [32000,512]
  const float* pos    = (const float*)d_in[4];   // [1024,512]
  auto LP = [&](int l, int j) { return (const float*)d_in[5 + l * 16 + j]; };
  // j: 0 wq 1 bq 2 wk 3 bk 4 wv 5 bv 6 wo 7 bo 8 ln1g 9 ln1b
  //    10 w1 11 b1 12 w2 13 b2 14 ln2g 15 ln2b

  // ---- workspace layout ----
  char* ws = (char*)d_ws;
  size_t off = 0;
  auto wsalloc = [&](size_t bytes) -> void* {
    void* p = ws + off;
    off += (bytes + 255) & ~(size_t)255;
    return p;
  };
  const size_t actB = (size_t)NTOK * DMODEL * sizeof(float);
  float*     X     = (float*)wsalloc(actB);
  float*     Q     = (float*)wsalloc(actB);
  float*     Kb    = (float*)wsalloc(actB);
  float*     V     = (float*)wsalloc(actB);
  float*     T4    = (float*)wsalloc(actB);
  const int  CHUNK = 8192;
  float*     Hbuf  = (float*)wsalloc((size_t)CHUNK * DFF_N * sizeof(float));
  _Float16*  W16   = (_Float16*)wsalloc((size_t)2 * 3145728 * sizeof(_Float16));
  float*     maskf = (float*)wsalloc((size_t)NTOK * sizeof(float));
  float*     ctxU  = (float*)wsalloc((size_t)16 * DMODEL * sizeof(float));
  float*     knowU = (float*)wsalloc((size_t)512 * DMODEL * sizeof(float));

  // ---- 1. convert weights to f16 (stay hot in 192MB L2) ----
  const int wIdx[6]  = {0, 2, 4, 6, 10, 12};                 // wq wk wv wo w1 w2
  const int wCnt[6]  = {262144, 262144, 262144, 262144, 1048576, 1048576};
  const size_t wOff[6] = {0, 262144, 524288, 786432, 1048576, 2097152};
  for (int l = 0; l < 2; ++l)
    for (int j = 0; j < 6; ++j) {
      int n = wCnt[j];
      f32_to_f16_kernel<<<(n + 255) / 256, 256, 0, stream>>>(
          LP(l, wIdx[j]), W16 + (size_t)l * 3145728 + wOff[j], n);
    }

  // ---- 2. embedding + positions + pad masks ----
  embed_kernel<<<NTOK, 256, 0, stream>>>(src, know, emb, pos, X, maskf);

  // ---- 3. encoder layers ----
  auto gemm = [&](const float* A, const _Float16* W, const float* bias, float* C,
                  int M, int N, int Kdim, int relu) {
    dim3 grid(M / BM, N / BN);
    gemm_wmma_f16<<<grid, 256, 0, stream>>>(A, W, bias, C, M, N, Kdim, relu);
  };

  for (int l = 0; l < 2; ++l) {
    const _Float16* Wl = W16 + (size_t)l * 3145728;
    // Q/K/V projections
    gemm(X, Wl + wOff[0], LP(l, 1), Q,  NTOK, DMODEL, DMODEL, 0);
    gemm(X, Wl + wOff[1], LP(l, 3), Kb, NTOK, DMODEL, DMODEL, 0);
    gemm(X, Wl + wOff[2], LP(l, 5), V,  NTOK, DMODEL, DMODEL, 0);
    // attention: ctx (16 seqs of 256) and knowledge (512 seqs of 128)
    attn_kernel<<<16 * 8 * 256, 64, 0, stream>>>(Q, Kb, V, maskf, T4, 256, 0);
    attn_kernel<<<512 * 8 * 128, 64, 0, stream>>>(Q, Kb, V, maskf, T4, 128, CTX_ROWS);
    // output projection (reuse Q as destination)
    gemm(T4, Wl + wOff[3], LP(l, 7), Q, NTOK, DMODEL, DMODEL, 0);
    // x = LN(x + attn)
    ln_residual_kernel<<<NTOK, 64, 0, stream>>>(X, Q, LP(l, 8), LP(l, 9));
    // FFN, chunked so the 2048-wide hidden stays at 64MB
    for (int c0 = 0; c0 < NTOK; c0 += CHUNK) {
      int M = (NTOK - c0 < CHUNK) ? (NTOK - c0) : CHUNK;
      gemm(X + (size_t)c0 * DMODEL, Wl + wOff[4], LP(l, 11), Hbuf, M, DFF_N, DMODEL, 1);
      gemm(Hbuf, Wl + wOff[5], LP(l, 13), T4 + (size_t)c0 * DMODEL, M, DMODEL, DFF_N, 0);
    }
    // x = LN(x + ffn)
    ln_residual_kernel<<<NTOK, 64, 0, stream>>>(X, T4, LP(l, 14), LP(l, 15));
  }

  // ---- 4. sentence embeddings + knowledge scores ----
  use_emb_kernel<<<16, 128, 0, stream>>>(X, maskf, ctxU, 256, 0);
  use_emb_kernel<<<512, 128, 0, stream>>>(X, maskf, knowU, 128, CTX_ROWS);

  float* out = (float*)d_out;
  const size_t MASK_OFF = (size_t)16 * 384 * DMODEL;   // 3,145,728
  const size_t CK_OFF   = MASK_OFF + (size_t)16 * 384; // 3,151,872
  ck_attn_kernel<<<512, 64, 0, stream>>>(ctxU, knowU, out + CK_OFF);

  // ---- 5. assemble full_enc + full_mask ----
  assemble_kernel<<<16 * 384, 256, 0, stream>>>(X, maskf, chosen,
                                                out, out + MASK_OFF);
}
